// GAT_87144886436203
// MI455X (gfx1250) — matrix operand order repaired and checked by reading, not compiled, not run on hardware
//
#include <hip/hip_runtime.h>

// ---------------------------------------------------------------------------
// GAT on gfx1250 (MI455X): bf16 WMMA GEMMs + fused masked-softmax aggregation
// + Tensor Data Mover (TDM) staging for linear/2-D tiles
// ---------------------------------------------------------------------------
#define Nn    4096
#define FEAT  512      // == HEADS*HID, so layer-2 GEMM reuses the same kernel
#define HID   64
#define HEADS 8
#define LRA   0.2f     // LeakyReLU slope

typedef __attribute__((ext_vector_type(16))) __bf16 v16bf;
typedef __attribute__((ext_vector_type(8)))  float  v8f;
typedef __attribute__((ext_vector_type(4)))  unsigned int v4u;
typedef __attribute__((ext_vector_type(8)))  int  v8i_;
typedef __attribute__((ext_vector_type(4)))  int  v4i_;

union BFrag { v16bf v; unsigned int u[8]; };

__device__ __forceinline__ unsigned short f2bf(float x) {           // RNE f32->bf16
  unsigned int b = __float_as_uint(x);
  b += 0x7FFFu + ((b >> 16) & 1u);
  return (unsigned short)(b >> 16);
}
__device__ __forceinline__ unsigned short f2bf_rz(float x) {        // truncate (hot path)
  return (unsigned short)(__float_as_uint(x) >> 16);
}
__device__ __forceinline__ float bf2f(unsigned short h) {
  return __uint_as_float(((unsigned int)h) << 16);
}
// order-preserving float<->uint map for atomicMax on signed floats
__device__ __forceinline__ unsigned int fmapu(float x) {
  unsigned int b = __float_as_uint(x);
  return (b & 0x80000000u) ? ~b : (b | 0x80000000u);
}
__device__ __forceinline__ float funmap(unsigned int u) {
  unsigned int b = (u & 0x80000000u) ? (u & 0x7FFFFFFFu) : ~u;
  return __uint_as_float(b);
}

// --- TDM: issue one tensor_load_to_lds for a 2-D tile (ISA 8.3/8.4 D#) -----
// data_size_code: 0=1B 1=2B 2=4B. Dims/strides in elements.
// This toolchain's builtin is the 6-arg form:
//   (uint32x4 g0, int32x8 g1, int32x4 g2, int32x4 g3, int32x8 extra, i32 cpol)
__device__ __forceinline__ void tdm_load_2d(void* lds_dst, const void* gsrc,
                                            unsigned int data_size_code,
                                            unsigned int tensor_d0,
                                            unsigned long long stride0,
                                            unsigned int tensor_d1,
                                            unsigned int tile_d0,
                                            unsigned int tile_d1) {
  unsigned long long ga = (unsigned long long)gsrc;
  v4u g0;
  g0[0] = 1u;                                         // count=1 (valid), user mode
  g0[1] = (unsigned int)(size_t)lds_dst;              // LDS addr = flat[31:0]
  g0[2] = (unsigned int)ga;                           // global_addr[31:0]
  g0[3] = (unsigned int)((ga >> 32) & 0x01FFFFFFull)  // global_addr[56:32]
          | (2u << 30);                               // type = 2 ("image")
  v8i_ g1;
  g1[0] = (int)(data_size_code << 16);                // wg_mask=0, data_size
  g1[1] = (int)((tensor_d0 & 0xFFFFu) << 16);         // tensor_dim0[15:0]
  g1[2] = (int)((tensor_d0 >> 16) | ((tensor_d1 & 0xFFFFu) << 16));
  g1[3] = (int)((tensor_d1 >> 16) | (tile_d0 << 16)); // tile_dim0
  g1[4] = (int)(tile_d1 & 0xFFFFu);                   // tile_dim1 (tile_dim2=0)
  g1[5] = (int)(unsigned int)(stride0 & 0xFFFFFFFFull);     // dim0_stride[31:0]
  g1[6] = (int)(unsigned int)((stride0 >> 32) & 0xFFFFull); // dim0_stride[47:32]
  g1[7] = 0;
  v4i_ z4 = {0, 0, 0, 0};
  v8i_ z8 = {0, 0, 0, 0, 0, 0, 0, 0};
  __builtin_amdgcn_tensor_load_to_lds(g0, g1, z4, z4, z8, 0);
}

// --- K0: x (f32) -> bf16, init gmax slots ----------------------------------
__global__ void k0_convert(const float* __restrict__ x,
                           unsigned short* __restrict__ xb,
                           unsigned int* __restrict__ gmaxu) {
  int i = (blockIdx.x * 256 + threadIdx.x) * 4;
  float4 v = *(const float4*)(x + i);
  uint2 o;
  o.x = (unsigned int)f2bf(v.x) | ((unsigned int)f2bf(v.y) << 16);
  o.y = (unsigned int)f2bf(v.z) | ((unsigned int)f2bf(v.w) << 16);
  *(uint2*)(xb + i) = o;
  if (blockIdx.x == 0 && threadIdx.x < 16) gmaxu[threadIdx.x] = 0u;
}

// --- K1: adjacency int32 -> bitmask (wave32 ballot) ------------------------
__global__ void k1_pack(const int* __restrict__ adj,
                        unsigned int* __restrict__ maskb) {
  int i = blockIdx.x * 256 + threadIdx.x;                 // 0 .. 16M-1
  unsigned long long b = __ballot(adj[i] > 0);
  if ((threadIdx.x & 31) == 0) maskb[i >> 5] = (unsigned int)b;
}

// --- K2: bf16 GEMM  Wh[head] = A[4096x512] @ W[head][512x64] ---------------
// block = 256 thr (8 waves), tile = 128 rows x 64 cols, K-chunks of 32
// A chunk (128x32 bf16, 2-D tile) staged by the Tensor Data Mover.
__global__ void k2_gemm(const unsigned short* __restrict__ A,   // [Nn][FEAT] bf16
                        const float* __restrict__ W,            // [heads][FEAT][HID]
                        unsigned short* __restrict__ Wh) {      // [heads][Nn][HID] bf16
  __shared__ __align__(128) unsigned short As[128 * 32];  // A chunk, row-major [r][k]
  __shared__ __align__(128) unsigned short Bs[HID * 32];  // B chunk, transposed [n][k]
  const int t = threadIdx.x, lane = t & 31, wv = t >> 5;
  const int row0 = blockIdx.x * 128;
  const float* Wb = W + (size_t)blockIdx.y * FEAT * HID;

  v8f zero = {0.f,0.f,0.f,0.f,0.f,0.f,0.f,0.f};
  v8f acc[4] = {zero, zero, zero, zero};

  const int kA_hi = (lane >= 16) ? 8 : 0;      // unused; A-frag K offset is kB (line below)
  const int kB    = (lane >= 16) ? 16 : 0;     // B-frag K offset
  const int mrowL = wv * 16 + (lane & 15);

#pragma unroll 1
  for (int kc = 0; kc < FEAT; kc += 32) {
    if (t == 0)   // TDM: 2-D tile, 128 rows x 32 bf16, row stride FEAT elements
      tdm_load_2d(As, A + (size_t)row0 * FEAT + kc, /*2B*/1,
                  FEAT, FEAT, Nn, /*tile*/32, 128);
    { // stage B transposed: Bs[n][k] = bf16(W[kc+k][n])  (needs f32->bf16 cvt)
      int kl = t >> 3, n0 = (t & 7) * 8;
      const float4* wsrc = (const float4*)(Wb + (size_t)(kc + kl) * HID + n0);
      float4 a0 = wsrc[0], a1 = wsrc[1];
      Bs[(n0+0)*32 + kl] = f2bf(a0.x); Bs[(n0+1)*32 + kl] = f2bf(a0.y);
      Bs[(n0+2)*32 + kl] = f2bf(a0.z); Bs[(n0+3)*32 + kl] = f2bf(a0.w);
      Bs[(n0+4)*32 + kl] = f2bf(a1.x); Bs[(n0+5)*32 + kl] = f2bf(a1.y);
      Bs[(n0+6)*32 + kl] = f2bf(a1.z); Bs[(n0+7)*32 + kl] = f2bf(a1.w);
    }
    if (t == 0) __builtin_amdgcn_s_wait_tensorcnt(0);
    __syncthreads();
    // A fragment: lane l holds row (l&15); VGPR v -> K = 16*(l>=16)+2*v+p
    BFrag fa;
#pragma unroll
    for (int v = 0; v < 8; ++v) {
      int K = kB + 2 * v;
      fa.u[v] = *(const unsigned int*)(As + mrowL * 32 + K);
    }
#pragma unroll
    for (int c = 0; c < 4; ++c) {
      BFrag fb;                             // B: lane l -> N=(l&15)+16c, K=2v+p+kB
      int ncol = c * 16 + (lane & 15);
#pragma unroll
      for (int v = 0; v < 8; ++v)
        fb.u[v] = *(const unsigned int*)(Bs + ncol * 32 + kB + 2 * v);
      acc[c] = __builtin_amdgcn_wmma_f32_16x16x32_bf16(
          false, fa.v, false, fb.v, (short)0, acc[c], false, false);
    }
    __syncthreads();
  }
  // epilogue: C/D layout VGPR r, lane l -> M = r + 8*(l>=16), N = (l&15)+16c
  unsigned short* outp = Wh + (size_t)blockIdx.y * Nn * HID;
  const int mhi = (lane >= 16) ? 8 : 0;
#pragma unroll
  for (int c = 0; c < 4; ++c)
#pragma unroll
    for (int r = 0; r < 8; ++r) {
      int row = row0 + wv * 16 + r + mhi;
      int col = c * 16 + (lane & 15);
      outp[(size_t)row * HID + col] = f2bf(acc[c][r]);
    }
}

// --- K3: f = Wh@a_src, g = Wh@a_dst, gmax = max(g) per head ----------------
__global__ void k3_fg(const unsigned short* __restrict__ Wh,  // [heads][Nn][HID]
                      const float* __restrict__ aS, const float* __restrict__ aD,
                      float* __restrict__ f, float* __restrict__ g,
                      unsigned int* __restrict__ gmaxu) {
  const int head = blockIdx.y;
  __shared__ float s_a[2 * HID];
  if (threadIdx.x < 2 * HID)
    s_a[threadIdx.x] = (threadIdx.x < HID) ? aS[head * HID + threadIdx.x]
                                           : aD[head * HID + threadIdx.x - HID];
  __syncthreads();
  int row = blockIdx.x * 256 + threadIdx.x;
  const unsigned short* wr = Wh + ((size_t)head * Nn + row) * HID;
  float sf = 0.f, sg = 0.f;
#pragma unroll
  for (int c0 = 0; c0 < HID; c0 += 8) {
    uint4 v = *(const uint4*)(wr + c0);
    unsigned int w4[4] = {v.x, v.y, v.z, v.w};
#pragma unroll
    for (int q = 0; q < 4; ++q) {
      float lo = bf2f((unsigned short)(w4[q] & 0xFFFFu));
      float hi = bf2f((unsigned short)(w4[q] >> 16));
      sf += lo * s_a[c0 + 2*q]       + hi * s_a[c0 + 2*q + 1];
      sg += lo * s_a[HID + c0 + 2*q] + hi * s_a[HID + c0 + 2*q + 1];
    }
  }
  f[head * Nn + row] = sf;
  g[head * Nn + row] = sg;
  unsigned int u = fmapu(sg);
#pragma unroll
  for (int s = 16; s >= 1; s >>= 1) {
    unsigned int o = __shfl_xor(u, s, 32);
    u = (o > u) ? o : u;
  }
  if ((threadIdx.x & 31) == 0) atomicMax(&gmaxu[head], u);
}

// --- K4: fused masked softmax + aggregation:  out = elu(softmax(e)@Wh) -----
// block = 256 thr (8 waves), 16 rows/wave, j swept in chunks of 32 via WMMA.
// g vector staged by TDM; Wh chunks double-buffered (1 barrier/chunk).
__global__ void k4_attn(const unsigned int* __restrict__ maskb,  // [Nn][128] bits
                        const unsigned short* __restrict__ Wh,   // [heads][Nn][HID]
                        const float* __restrict__ f, const float* __restrict__ g,
                        const unsigned int* __restrict__ gmaxu,
                        void* __restrict__ outp, int out_stride,
                        int head_col_stride, int out_f32) {
  const int head = blockIdx.y;
  const int row0 = blockIdx.x * 128;
  const int t = threadIdx.x, lane = t & 31, wv = t >> 5;

  __shared__ __align__(128) float gs[Nn];                    // 16KB: g for this head
  __shared__ float fs[128];
  __shared__ __align__(128) unsigned short Bs[2][HID * 32];  // 2x4KB Wh chunk [n][j]

  const unsigned short* Whh = Wh + (size_t)head * Nn * HID;
  if (t == 0)   // TDM: linear 16KB copy of g into LDS
    tdm_load_2d(gs, g + (size_t)head * Nn, /*4B*/2, Nn, Nn, 1, /*tile*/Nn, 1);
  if (t < 128) fs[t] = f[(size_t)head * Nn + row0 + t];

  // cooperative transposed staging of one Wh chunk [32 j][64 n] -> Bs[b][n][j]
  auto stageB = [&](int b, int j0) {
    int jl = t >> 3, n0 = (t & 7) * 8;
    uint4 v = *(const uint4*)(Whh + (size_t)j0 * HID + t * 8);
    unsigned int w4[4] = {v.x, v.y, v.z, v.w};
#pragma unroll
    for (int q = 0; q < 4; ++q) {
      Bs[b][(n0 + 2*q + 0) * 32 + jl] = (unsigned short)(w4[q] & 0xFFFFu);
      Bs[b][(n0 + 2*q + 1) * 32 + jl] = (unsigned short)(w4[q] >> 16);
    }
  };
  stageB(0, 0);
  if (t == 0) __builtin_amdgcn_s_wait_tensorcnt(0);
  __syncthreads();

  const int mrowL = wv * 16 + (lane & 15);
  const unsigned int* mrow = maskb + (size_t)(row0 + mrowL) * 128;
  const float freg = fs[mrowL];
  const float gmax = funmap(gmaxu[head]);
  float mval = freg + gmax;                       // monotone lrelu => row max bound
  mval = (mval > 0.f) ? mval : LRA * mval;

  v8f zero = {0.f,0.f,0.f,0.f,0.f,0.f,0.f,0.f};
  v8f acc[4] = {zero, zero, zero, zero};
  float lsum = 0.f;
  const int kA_hi = (lane >= 16) ? 8 : 0;
  const int kB    = (lane >= 16) ? 16 : 0;

#pragma unroll 1
  for (int j0 = 0; j0 < Nn; j0 += 32) {
    const int cur = (j0 >> 5) & 1;
    if (j0 + 32 < Nn) {                          // prefetch next chunk into other buf
      stageB(cur ^ 1, j0 + 32);
      __builtin_prefetch(Whh + (size_t)(j0 + 64) * HID, 0, 1);
    }
    const unsigned int mw = mrow[j0 >> 5];       // 32 adjacency bits for this row
    BFrag fa;                                    // p_ij tile in WMMA A layout
#pragma unroll
    for (int v = 0; v < 8; ++v) {
      unsigned int pk = 0;
#pragma unroll
      for (int p = 0; p < 2; ++p) {
        int K = kB + 2 * v + p;
        float e = freg + gs[j0 + K];
        e = (e > 0.f) ? e : LRA * e;
        float pv = ((mw >> K) & 1u) ? __expf(e - mval) : 0.f;
        lsum += pv;
        pk |= ((unsigned int)f2bf_rz(pv)) << (16 * p);
      }
      fa.u[v] = pk;
    }
    const unsigned short* Bc = Bs[cur];
#pragma unroll
    for (int c = 0; c < 4; ++c) {
      BFrag fb;
      int ncol = c * 16 + (lane & 15);
#pragma unroll
      for (int v = 0; v < 8; ++v)
        fb.u[v] = *(const unsigned int*)(Bc + ncol * 32 + kB + 2 * v);
      acc[c] = __builtin_amdgcn_wmma_f32_16x16x32_bf16(
          false, fa.v, false, fb.v, (short)0, acc[c], false, false);
    }
    __syncthreads();                             // staged buf visible, reads done
  }

  // row sums: lanes l and l+16 cover disjoint K of the same row
  float ls = lsum + __shfl_xor(lsum, 16, 32);    // lane q holds sum of row (q&15)
  const int mhi = (lane >= 16) ? 8 : 0;
  float inv[8];
#pragma unroll
  for (int r = 0; r < 8; ++r) inv[r] = 1.0f / __shfl(ls, r + mhi, 32);

  const int colbase = head * head_col_stride + (lane & 15);
#pragma unroll
  for (int c = 0; c < 4; ++c)
#pragma unroll
    for (int r = 0; r < 8; ++r) {
      float v = acc[c][r] * inv[r];
      v = (v > 0.f) ? v : (__expf(v) - 1.0f);    // ELU
      int row = row0 + wv * 16 + r + mhi;
      int col = colbase + 16 * c;
      if (out_f32) ((float*)outp)[(size_t)row * out_stride + col] = v;
      else ((unsigned short*)outp)[(size_t)row * out_stride + col] = f2bf(v);
    }
}

// ---------------------------------------------------------------------------
extern "C" void kernel_launch(void* const* d_in, const int* in_sizes, int n_in,
                              void* d_out, int out_size, void* d_ws, size_t ws_size,
                              hipStream_t stream) {
  const float* x         = (const float*)d_in[0];
  const int*   adj       = (const int*)d_in[1];
  const float* W_heads   = (const float*)d_in[2];
  const float* a_src     = (const float*)d_in[3];
  const float* a_dst     = (const float*)d_in[4];
  const float* W_out     = (const float*)d_in[5];
  const float* a_src_out = (const float*)d_in[6];
  const float* a_dst_out = (const float*)d_in[7];
  float* out = (float*)d_out;

  char* ws = (char*)d_ws;
  size_t off = 0;
  auto take = [&](size_t bytes) -> char* {
    char* p = ws + off;
    off = (off + bytes + 255) & ~(size_t)255;
    return p;
  };
  unsigned short* xb    = (unsigned short*)take((size_t)Nn * FEAT * 2);     // 4 MB
  unsigned int*   maskb = (unsigned int*)  take((size_t)Nn * 128 * 4);      // 2 MB
  unsigned short* Wh1   = (unsigned short*)take((size_t)HEADS * Nn * HID*2);// 4 MB
  float*          f1    = (float*)take((size_t)HEADS * Nn * 4);
  float*          g1    = (float*)take((size_t)HEADS * Nn * 4);
  unsigned short* hcat  = (unsigned short*)take((size_t)Nn * FEAT * 2);     // 4 MB
  unsigned short* Wh2   = (unsigned short*)take((size_t)Nn * HID * 2);
  float*          f2    = (float*)take((size_t)Nn * 4);
  float*          g2    = (float*)take((size_t)Nn * 4);
  unsigned int*   gmaxu = (unsigned int*)take(16 * 4);
  (void)ws_size; (void)in_sizes; (void)n_in; (void)out_size;

  // x -> bf16, init gmax
  k0_convert<<<(Nn * FEAT) / 1024, 256, 0, stream>>>(x, xb, gmaxu);
  // adj -> bitmask (L2-resident thereafter)
  k1_pack<<<(Nn * Nn) / 256, 256, 0, stream>>>(adj, maskb);
  // layer 1
  k2_gemm<<<dim3(Nn / 128, HEADS), 256, 0, stream>>>(xb, W_heads, Wh1);
  k3_fg  <<<dim3(Nn / 256, HEADS), 256, 0, stream>>>(Wh1, a_src, a_dst, f1, g1, gmaxu);
  k4_attn<<<dim3(Nn / 128, HEADS), 256, 0, stream>>>(maskb, Wh1, f1, g1, gmaxu,
                                                     hcat, FEAT, HID, 0);
  // layer 2 (same shapes: FEAT=HEADS*HID, HID=NCLASS)
  k2_gemm<<<dim3(Nn / 128, 1), 256, 0, stream>>>(hcat, W_out, Wh2);
  k3_fg  <<<dim3(Nn / 256, 1), 256, 0, stream>>>(Wh2, a_src_out, a_dst_out, f2, g2,
                                                 gmaxu + 8);
  k4_attn<<<dim3(Nn / 128, 1), 256, 0, stream>>>(maskb, Wh2, f2, g2, gmaxu + 8,
                                                 out, HID, 0, 1);
}
